// STGN_C_74363063763509
// MI455X (gfx1250) — matrix operand order repaired and checked by reading, not compiled, not run on hardware
//
#include <hip/hip_runtime.h>

#define T_STEPS 8
#define N_NODES 10000
#define IN_DIM 64
#define HID_DIM 128
#define N_EDGES 320000
#define M_ROWS (T_STEPS * N_NODES)

typedef __attribute__((ext_vector_type(16))) __bf16    v16bf;
typedef __attribute__((ext_vector_type(8)))  __bf16    v8bf;
typedef __attribute__((ext_vector_type(2)))  __bf16    bf16x2;
typedef __attribute__((ext_vector_type(16))) float     v16f;
typedef __attribute__((ext_vector_type(8)))  float     v8f;
typedef __attribute__((ext_vector_type(2)))  float     f32x2;
typedef __attribute__((ext_vector_type(8)))  unsigned  v8u;
typedef __attribute__((ext_vector_type(4)))  unsigned  v4u;

// pack two f32 into one dword of two bf16 via vector fptrunc (native cvt when available)
__device__ __forceinline__ unsigned pk2(float x, float y) {
  f32x2 f = {x, y};
  return __builtin_bit_cast(unsigned, __builtin_convertvector(f, bf16x2));
}

__global__ void k_zero(float* __restrict__ p, int n) {
  int i = blockIdx.x * blockDim.x + threadIdx.x;
  if (i < n) p[i] = 0.0f;
}

// deg[dst] += ew  (gcn_norm, add_self_loops=False)
__global__ void k_deg(const int* __restrict__ dst, const float* __restrict__ ew,
                      float* __restrict__ deg) {
  int e = blockIdx.x * blockDim.x + threadIdx.x;
  if (e < N_EDGES) atomicAdd(&deg[dst[e]], ew[e]);
}

// norm[e] = dis[src]*ew*dis[dst]; outw[src] += norm[e] (readout-absorbed layer-2 scatter)
__global__ void k_norm(const int* __restrict__ src, const int* __restrict__ dst,
                       const float* __restrict__ ew, const float* __restrict__ deg,
                       float* __restrict__ nrm, float* __restrict__ outw) {
  int e = blockIdx.x * blockDim.x + threadIdx.x;
  if (e >= N_EDGES) return;
  int s = src[e], d = dst[e];
  float ds_ = deg[s], dd = deg[d];
  float is = ds_ > 0.f ? rsqrtf(ds_) : 0.f;
  float id = dd > 0.f ? rsqrtf(dd) : 0.f;
  float w = is * ew[e] * id;
  nrm[e] = w;
  atomicAdd(&outw[s], w);
}

// outwX[row] = outw[row % N]  (precompute so the GEMM2 epilogue is two float4 loads)
__global__ void k_expand(const float* __restrict__ outw, float* __restrict__ outwX) {
  int r = blockIdx.x * blockDim.x + threadIdx.x;
  if (r < M_ROWS) outwX[r] = outw[r % N_NODES];
}

// ax[t,dst,:] += norm[e] * x[t,src,:]   (64-wide, pre-GEMM aggregation; L2-resident)
// 16 threads per edge, each owns 4 consecutive features, loops over T.
__global__ void k_aggregate(const float* __restrict__ x, const int* __restrict__ src,
                            const int* __restrict__ dst, const float* __restrict__ nrm,
                            float* __restrict__ ax) {
  int gid = blockIdx.x * blockDim.x + threadIdx.x;
  int e = gid >> 4;
  if (e >= N_EDGES) return;
  int q = (gid & 15) << 2;
  float w = nrm[e];
  if (w == 0.f) return;
  int s = src[e], d = dst[e];
  const float* xs = x + (size_t)s * IN_DIM + q;
  float* ad = ax + (size_t)d * IN_DIM + q;
  __builtin_prefetch(xs, 0, 1);                     // global_prefetch_b8
  const size_t step = (size_t)N_NODES * IN_DIM;
#pragma unroll
  for (int t = 0; t < T_STEPS; ++t) {
    float4 v = *(const float4*)(xs + (size_t)t * step);
    float* o = ad + (size_t)t * step;
    atomicAdd(o + 0, w * v.x);
    atomicAdd(o + 1, w * v.y);
    atomicAdd(o + 2, w * v.z);
    atomicAdd(o + 3, w * v.w);
  }
}

// y1 = relu(ax @ W1 + b1)   M=80000, K=64, N=128.  One 16x16 tile per wave,
// 8 waves/block cover all 8 N-tiles of one M-tile. W1 staged transposed bf16 in LDS,
// output tile staged in LDS for coalesced b128 stores.
__global__ __launch_bounds__(256) void k_gemm1(const float* __restrict__ ax,
                                               const float* __restrict__ W1,
                                               const float* __restrict__ b1,
                                               __bf16* __restrict__ y1) {
  __shared__ __align__(16) unsigned sWu[HID_DIM * IN_DIM / 2];   // [n][k/2] 16KB
  __shared__ __align__(16) __bf16   sOut[16 * HID_DIM];          // 4KB D tile
  int tid = threadIdx.x;
  // coalesced stage: pair p=k/2, read two full 128-wide rows of W1
  for (int idx = tid; idx < (IN_DIM / 2) * HID_DIM; idx += 256) {
    int p = idx >> 7, n = idx & 127;           // p: 0..31, n: 0..127
    sWu[n * (IN_DIM / 2) + p] = pk2(W1[(2 * p) * HID_DIM + n],
                                    W1[(2 * p + 1) * HID_DIM + n]);
  }
  __syncthreads();

  int wave = tid >> 5, lane = tid & 31;
  int khalf = lane >> 4, lcol = lane & 15;
  int mbase = blockIdx.x * 16;
  int ncol = wave * 16 + lcol;

  v8f c = {};
  const float* rp = ax + (size_t)(mbase + lcol) * IN_DIM;
#pragma unroll
  for (int ks = 0; ks < 2; ++ks) {
    // A 16x32 bf16 layout: lane<16 -> K {0..7,16..23}, lane>=16 -> K {8..15,24..31}
    int k0 = ks * 32 + khalf * 8;
    float4 a0 = *(const float4*)(rp + k0);
    float4 a1 = *(const float4*)(rp + k0 + 4);
    float4 a2 = *(const float4*)(rp + k0 + 16);
    float4 a3 = *(const float4*)(rp + k0 + 20);
    v16f af = { a0.x, a0.y, a0.z, a0.w, a1.x, a1.y, a1.z, a1.w,
                a2.x, a2.y, a2.z, a2.w, a3.x, a3.y, a3.z, a3.w };
    v16bf A = __builtin_convertvector(af, v16bf);
    // B 32x16: lanes 0-15 hold K {0..7,16..23}, lanes 16-31 hold K {8..15,24..31} of this k-step
    const unsigned* bp = sWu + ncol * (IN_DIM / 2) + ks * 16 + khalf * 4;
    v4u blo = *(const v4u*)bp;
    v4u bhi = *(const v4u*)(bp + 8);
    v8u bu = { blo[0], blo[1], blo[2], blo[3], bhi[0], bhi[1], bhi[2], bhi[3] };
    v16bf B = __builtin_bit_cast(v16bf, bu);
    c = __builtin_amdgcn_wmma_f32_16x16x32_bf16(false, A, false, B, (short)0, c,
                                                false, false);
  }
  float bias = b1[ncol];
  v8f cb;
#pragma unroll
  for (int i = 0; i < 8; ++i) cb[i] = fmaxf(c[i] + bias, 0.f);   // relu = v_max
  v8bf cv = __builtin_convertvector(cb, v8bf);
#pragma unroll
  for (int i = 0; i < 8; ++i)                   // C/D: vgpr i -> row i + 8*khalf
    sOut[(khalf * 8 + i) * HID_DIM + ncol] = cv[i];
  __syncthreads();
  // coalesced write-out: 16 rows x 128 cols bf16 = 4KB; 256 lanes x 16B
  int r = tid >> 4, cg = (tid & 15) * 8;
  *(v8bf*)(y1 + (size_t)(mbase + r) * HID_DIM + cg) =
      *(const v8bf*)(sOut + r * HID_DIM + cg);
}

// h2 = y1 @ W2 (M=80000, K=128, N=64) with fused readout:
// acc[n] += (1/TN) * sum_rows outwX[row] * h2[row][n]   (no second scatter)
__global__ __launch_bounds__(128) void k_gemm2(const __bf16* __restrict__ y1,
                                               const float* __restrict__ W2,
                                               const float* __restrict__ outwX,
                                               float* __restrict__ acc) {
  __shared__ __align__(16) unsigned sWu[IN_DIM * HID_DIM / 2];   // [n][k/2] 16KB
  int tid = threadIdx.x;
  for (int idx = tid; idx < (HID_DIM / 2) * IN_DIM; idx += 128) {
    int p = idx >> 6, n = idx & 63;            // p: 0..63, n: 0..63
    sWu[n * (HID_DIM / 2) + p] = pk2(W2[(2 * p) * IN_DIM + n],
                                     W2[(2 * p + 1) * IN_DIM + n]);
  }
  __syncthreads();

  int wave = tid >> 5, lane = tid & 31;
  int khalf = lane >> 4, lcol = lane & 15;
  int mbase = blockIdx.x * 16;
  int ncol = wave * 16 + lcol;

  v8f c = {};
  const __bf16* rp = y1 + (size_t)(mbase + lcol) * HID_DIM;
#pragma unroll
  for (int ks = 0; ks < 4; ++ks) {
    int k0 = ks * 32 + khalf * 8;
    v4u alo = *(const v4u*)(rp + k0);
    v4u ahi = *(const v4u*)(rp + k0 + 16);
    v8u au = { alo[0], alo[1], alo[2], alo[3], ahi[0], ahi[1], ahi[2], ahi[3] };
    v16bf A = __builtin_bit_cast(v16bf, au);
    const unsigned* bp = sWu + ncol * (HID_DIM / 2) + ks * 16 + khalf * 4;
    v4u blo = *(const v4u*)bp;
    v4u bhi = *(const v4u*)(bp + 8);
    v8u bu = { blo[0], blo[1], blo[2], blo[3], bhi[0], bhi[1], bhi[2], bhi[3] };
    v16bf B = __builtin_bit_cast(v16bf, bu);
    c = __builtin_amdgcn_wmma_f32_16x16x32_bf16(false, A, false, B, (short)0, c,
                                                false, false);
  }
  const float invTN = 1.0f / (float)M_ROWS;
  const float4* wp = (const float4*)(outwX + mbase + khalf * 8);  // rows of this half
  float4 w0 = wp[0], w1 = wp[1];
  float partial = c[0] * w0.x + c[1] * w0.y + c[2] * w0.z + c[3] * w0.w +
                  c[4] * w1.x + c[5] * w1.y + c[6] * w1.z + c[7] * w1.w;
  atomicAdd(&acc[ncol], partial * invTN);
}

// m_v = acc_v + b2 ; z = normalize(m) ; degenerate NTXent computed literally
__global__ void k_finalize(const float* __restrict__ acc, const float* __restrict__ b2,
                           float* __restrict__ out) {
  __shared__ float m1[64], m2[64];
  int i = threadIdx.x;
  m1[i] = acc[i] + b2[i];
  m2[i] = acc[64 + i] + b2[i];
  __syncthreads();
  if (i == 0) {
    float s1 = 0.f, s2 = 0.f;
    for (int j = 0; j < 64; ++j) { s1 += m1[j] * m1[j]; s2 += m2[j] * m2[j]; }
    float n1 = fmaxf(sqrtf(s1), 1e-12f), n2 = fmaxf(sqrtf(s2), 1e-12f);
    const float tiny = 1.1754943508222875e-38f;
    float loss = 0.f;
    for (int j = 0; j < 64; ++j) {
      float z1 = m1[j] / n1, z2 = m2[j] / n2;
      float e1 = z1 / fmaxf(fabsf(z1), 1e-12f);
      float e2 = z2 / fmaxf(fabsf(z2), 1e-12f);
      float pos = (e1 * e2) * 10.0f;            // /TEMP
      float num = __expf(pos - pos);            // max_val == pos (negs masked)
      float den = num;                          // sum(exp(-inf)) == 0
      loss += -__logf(num / den + tiny);
    }
    out[0] = loss * (1.0f / 64.0f);
  }
}

extern "C" void kernel_launch(void* const* d_in, const int* in_sizes, int n_in,
                              void* d_out, int out_size, void* d_ws, size_t ws_size,
                              hipStream_t stream) {
  const float* view1 = (const float*)d_in[0];
  const float* view2 = (const float*)d_in[1];
  const int*   eidx  = (const int*)d_in[2];   // [2, E]
  const float* ew    = (const float*)d_in[3];
  const float* W1    = (const float*)d_in[4];
  const float* b1    = (const float*)d_in[5];
  const float* W2    = (const float*)d_in[6];
  const float* b2    = (const float*)d_in[7];
  const int* src = eidx;
  const int* dst = eidx + N_EDGES;

  float*  deg   = (float*)d_ws;
  float*  outw  = deg + N_NODES;
  float*  acc   = outw + N_NODES;                                   // 128 f32
  float*  outwX = acc + 128;                                        // M_ROWS f32
  float*  nrm   = outwX + M_ROWS;                                   // E f32
  float*  ax    = nrm + N_EDGES;                                    // T*N*64 f32
  __bf16* y1    = (__bf16*)(ax + (size_t)M_ROWS * IN_DIM);          // T*N*128 bf16

  const int zn = 2 * N_NODES + 128;
  const int axN = M_ROWS * IN_DIM;
  const int Mtiles = M_ROWS / 16;               // 5000

  k_zero<<<(zn + 255) / 256, 256, 0, stream>>>(deg, zn);
  k_deg<<<(N_EDGES + 255) / 256, 256, 0, stream>>>(dst, ew, deg);
  k_norm<<<(N_EDGES + 255) / 256, 256, 0, stream>>>(src, dst, ew, deg, nrm, outw);
  k_expand<<<(M_ROWS + 255) / 256, 256, 0, stream>>>(outw, outwX);

  for (int v = 0; v < 2; ++v) {
    const float* xv = v ? view2 : view1;
    k_zero<<<(axN + 255) / 256, 256, 0, stream>>>(ax, axN);
    k_aggregate<<<(N_EDGES * 16 + 255) / 256, 256, 0, stream>>>(xv, src, dst, nrm, ax);
    k_gemm1<<<Mtiles, 256, 0, stream>>>(ax, W1, b1, y1);
    k_gemm2<<<Mtiles, 128, 0, stream>>>(y1, W2, outwX, acc + v * 64);
  }
  k_finalize<<<1, 64, 0, stream>>>(acc, b2, (float*)d_out);

  (void)in_sizes; (void)n_in; (void)out_size; (void)ws_size;
}
